// S62D_18124761989177
// MI455X (gfx1250) — matrix operand order repaired and checked by reading, not compiled, not run on hardware
//
#include <hip/hip_runtime.h>

// ---------------- CDNA5 WMMA types ----------------
typedef __attribute__((ext_vector_type(16))) __bf16 v16bf;
typedef __attribute__((ext_vector_type(8)))  float  v8f;

#define D_MODEL 192
#define D_STATE 16
#define D_INNER 384
#define DT_RANK 12
#define KDIRS   4
#define Bb      4
#define Hh      64
#define Ww      64
#define Ll      4096            // H*W
#define Mtot    16384           // B*L
#define CHUNK   64              // scan timesteps staged per LDS chunk

// Map scan position l of direction k to canonical (h*W+w) index.
static __device__ __forceinline__ int perm_idx(int k, int l) {
    int ll = (k & 2) ? (Ll - 1 - l) : l;
    if (k & 1) { int h = ll & 63; int w = ll >> 6; return h * Ww + w; }
    return ll;
}

// A-fragment (16x32 bf16, MxK) from row-major f32 matrix, leading dim = lds.
// Contiguous k order used here: lane 0-15 -> M=lane; halves i: K = i + 16*(lane>>4)
static __device__ __forceinline__ v16bf load_a_bf16(const float* __restrict__ base,
                                                    int lds, int m0, int k0) {
    const int lane = threadIdx.x & 31;
    const int m    = m0 + (lane & 15);
    const int hi8  = (lane >> 4) << 4;
    const float* row = base + (long)m * lds;
    v16bf a;
#pragma unroll
    for (int i = 0; i < 16; ++i) {
        int k = k0 + i + hi8;
        a[i] = (__bf16)row[k];
    }
    return a;
}

// B-fragment (32x16 bf16, KxN); element (k,n) at base[k*sk + n*sn].
// Contiguous k order used here (matches load_a_bf16): lanes 0-15 hold K=k0..k0+15 (halves sequential), lanes 16-31 K=k0+16..k0+31.
static __device__ __forceinline__ v16bf load_b_bf16(const float* __restrict__ base,
                                                    int sk, int sn, int k0, int n0) {
    const int lane  = threadIdx.x & 31;
    const int n     = n0 + (lane & 15);
    const int kbase = k0 + ((lane >> 4) << 4);
    v16bf b;
#pragma unroll
    for (int i = 0; i < 16; ++i)
        b[i] = (__bf16)base[(long)(kbase + i) * sk + (long)n * sn];
    return b;
}

// ---------------- Kernel 0: pack x_proj_weight into zero-padded 48x384 panels ----
__global__ void pack_xpw(const float* __restrict__ xpw, float* __restrict__ xpw_pad) {
    const int idx = blockIdx.x * 256 + threadIdx.x;      // KDIRS*48*384 = 73728
    if (idx >= KDIRS * 48 * D_INNER) return;
    const int k   = idx / (48 * D_INNER);
    const int rem = idx % (48 * D_INNER);
    const int row = rem / D_INNER, col = rem % D_INNER;
    xpw_pad[idx] = (row < 44) ? xpw[((long)k * 44 + row) * D_INNER + col] : 0.0f;
}

// ---------------- Kernel 1: in-projection GEMM ----------------
// xz[m,e] = sum_c x[m,c] * Win[e,c];  e<384 -> xin[m,e] (l-major), else -> zbuf[m,e-384]
__global__ void gemm_inproj(const float* __restrict__ x, const float* __restrict__ win,
                            float* __restrict__ xin, float* __restrict__ zbuf) {
    const int wid = (blockIdx.x << 3) + (threadIdx.x >> 5);   // 1024*48 waves
    const int tn  = wid % 48;
    const int tm  = wid / 48;
    v8f c = {};
#pragma unroll
    for (int kk = 0; kk < D_MODEL; kk += 32) {
        v16bf a  = load_a_bf16(x, D_MODEL, tm * 16, kk);
        v16bf bf = load_b_bf16(win, 1, D_MODEL, kk, tn * 16);
        c = __builtin_amdgcn_wmma_f32_16x16x32_bf16(false, a, false, bf, (short)0, c, false, false);
    }
    const int lane = threadIdx.x & 31;
    const int n    = tn * 16 + (lane & 15);
    const int mh   = tm * 16 + ((lane >> 4) << 3);
#pragma unroll
    for (int j = 0; j < 8; ++j) {
        int m = mh + j;
        if (n < D_INNER) xin[(long)m * D_INNER + n]             = c[j];
        else             zbuf[(long)m * D_INNER + (n - D_INNER)] = c[j];
    }
}

// ---------------- Kernel 2: x_dbl GEMM per (b,k) ----------------
// x_dbl[bk,crow,l] = sum_d xpw_pad[k,crow,d] * xin[b, perm(k,l), d]
__global__ void gemm_xdbl(const float* __restrict__ xpw_pad, const float* __restrict__ xin,
                          float* __restrict__ dts_raw, float* __restrict__ Bsb,
                          float* __restrict__ Csb) {
    const int wid  = (blockIdx.x << 3) + (threadIdx.x >> 5); // 16*3*256 waves
    const int bk   = wid / 768;
    const int rem  = wid % 768;
    const int tm   = rem / 256;
    const int tn   = rem % 256;
    const int b    = bk >> 2, kdir = bk & 3;
    const int lane = threadIdx.x & 31;
    const int lcol = tn * 16 + (lane & 15);
    const int col  = perm_idx(kdir, lcol);
    const float* bcol = xin + ((long)b * Ll + col) * D_INNER;   // contiguous in d
    const float* arow = xpw_pad + (long)kdir * 48 * D_INNER;
    v8f c = {};
    for (int kk = 0; kk < D_INNER; kk += 32) {
        v16bf a = load_a_bf16(arow, D_INNER, tm * 16, kk);
        const int kbase = kk + ((lane >> 4) << 4);
        v16bf bf;
#pragma unroll
        for (int i = 0; i < 16; ++i)
            bf[i] = (__bf16)bcol[kbase + i];
        c = __builtin_amdgcn_wmma_f32_16x16x32_bf16(false, a, false, bf, (short)0, c, false, false);
    }
    const int mh = tm * 16 + ((lane >> 4) << 3);
#pragma unroll
    for (int j = 0; j < 8; ++j) {
        int m = mh + j;
        float v = c[j];
        if (m < DT_RANK)                    dts_raw[((long)bk * DT_RANK + m) * Ll + lcol] = v;
        else if (m < DT_RANK + D_STATE)     Bsb[((long)bk * D_STATE + (m - DT_RANK)) * Ll + lcol] = v;
        else if (m < DT_RANK + 2 * D_STATE) Csb[((long)bk * D_STATE + (m - DT_RANK - D_STATE)) * Ll + lcol] = v;
    }
}

// ---------------- Kernel 3: selective scan ----------------
// One block per (b,k); thread d owns state h[16].
// B/C/dt for CHUNK timesteps staged in LDS via async global->LDS copies (ASYNCcnt).
__global__ void scan_kernel(const float* __restrict__ xin, const float* __restrict__ dts_raw,
                            const float* __restrict__ Bsb, const float* __restrict__ Csb,
                            const float* __restrict__ dtw, const float* __restrict__ dtb,
                            const float* __restrict__ A_logs, const float* __restrict__ Dsv,
                            float* __restrict__ out_y) {
    const int bk   = blockIdx.x;          // 0..15
    const int b    = bk >> 2, kdir = bk & 3;
    const int d    = threadIdx.x;         // 0..383
    const int kd   = kdir * D_INNER + d;
    __shared__ float sBC[44][CHUNK];      // rows 0-15: B, 16-31: C, 32-43: dt_raw
    float w[DT_RANK], negA[D_STATE], h[D_STATE];
#pragma unroll
    for (int r = 0; r < DT_RANK; ++r) w[r] = dtw[(long)kd * DT_RANK + r];
#pragma unroll
    for (int n = 0; n < D_STATE; ++n) { negA[n] = -__expf(A_logs[(long)kd * D_STATE + n]); h[n] = 0.f; }
    const float bias = dtb[kd];
    const float Dd   = Dsv[kd];
    const float* xbase = xin + (long)b * Ll * D_INNER;
    float*       ybase = out_y + (long)bk * Ll * D_INNER;
    const float* srcB = Bsb     + (long)bk * D_STATE * Ll;
    const float* srcC = Csb     + (long)bk * D_STATE * Ll;
    const float* srcD = dts_raw + (long)bk * DT_RANK * Ll;

    for (int l0 = 0; l0 < Ll; l0 += CHUNK) {
        // ---- async stage 44 rows x CHUNK cols into LDS (b64 granules) ----
        for (int t = threadIdx.x; t < 44 * (CHUNK / 2); t += D_INNER) {
            const int r  = t >> 5;              // row 0..43
            const int cc = (t & 31) << 1;       // col 0,2,..,62
            const float* src = (r < 16) ? (srcB + (long)r * Ll)
                             : (r < 32) ? (srcC + (long)(r - 16) * Ll)
                                        : (srcD + (long)(r - 32) * Ll);
            unsigned long long gaddr = (unsigned long long)(const void*)(src + l0 + cc);
            unsigned ldsoff = (unsigned)(unsigned long long)(const void*)&sBC[r][cc];
            asm volatile("global_load_async_to_lds_b64 %0, %1, off"
                         :: "v"(ldsoff), "v"(gaddr) : "memory");
        }
        asm volatile("s_wait_asynccnt 0x0" ::: "memory");
        __syncthreads();

        // ---- CHUNK sequential steps ----
        for (int c = 0; c < CHUNK; ++c) {
            const int l   = l0 + c;
            const int idx = perm_idx(kdir, l);
            if (l + 8 < Ll)
                __builtin_prefetch(xbase + (long)perm_idx(kdir, l + 8) * D_INNER + d, 0, 1);
            const float u = xbase[(long)idx * D_INNER + d];
            float dtv = bias;
#pragma unroll
            for (int r = 0; r < DT_RANK; ++r) dtv += w[r] * sBC[32 + r][c];
            const float dt = (dtv > 20.f) ? dtv : log1pf(__expf(dtv));
            float y = 0.f;
#pragma unroll
            for (int n = 0; n < D_STATE; ++n) {
                h[n] = __expf(dt * negA[n]) * h[n] + dt * sBC[n][c] * u;
                y += h[n] * sBC[16 + n][c];
            }
            ybase[(long)idx * D_INNER + d] = y + u * Dd;   // canonical, coalesced over d
        }
        __syncthreads();
    }
}

// ---------------- Kernel 4: combine + LayerNorm + SiLU gate ----------------
__global__ void combine_ln_gate(const float* __restrict__ out_y, const float* __restrict__ zbuf,
                                const float* __restrict__ nw, const float* __restrict__ nb,
                                float* __restrict__ yg) {
    const int m = blockIdx.x;             // b*L + l
    const int b = m >> 12, l = m & (Ll - 1);
    const int d = threadIdx.x;            // 0..383
    float s = 0.f;
#pragma unroll
    for (int k = 0; k < KDIRS; ++k)
        s += out_y[((long)(b * KDIRS + k) * Ll + l) * D_INNER + d];

    float v1 = s, v2 = s * s;
#pragma unroll
    for (int off = 16; off > 0; off >>= 1) {
        v1 += __shfl_down(v1, off);
        v2 += __shfl_down(v2, off);
    }
    __shared__ float ps[12], pq[12];
    __shared__ float smu, srstd;
    const int wave = threadIdx.x >> 5;
    if ((threadIdx.x & 31) == 0) { ps[wave] = v1; pq[wave] = v2; }
    __syncthreads();
    if (threadIdx.x == 0) {
        float a = 0.f, q = 0.f;
        for (int i = 0; i < 12; ++i) { a += ps[i]; q += pq[i]; }
        float mu  = a * (1.f / D_INNER);
        float var = q * (1.f / D_INNER) - mu * mu;
        smu = mu; srstd = rsqrtf(var + 1e-6f);
    }
    __syncthreads();
    const float yn  = (s - smu) * srstd * nw[d] + nb[d];
    const float zv  = zbuf[(long)m * D_INNER + d];
    const float sil = zv / (1.f + __expf(-zv));
    yg[(long)m * D_INNER + d] = yn * sil;
}

// ---------------- Kernel 5: out-projection GEMM ----------------
// out[m,c] = sum_d yg[m,d] * Wout[c,d]
__global__ void gemm_out(const float* __restrict__ yg, const float* __restrict__ wout,
                         float* __restrict__ out) {
    const int wid = (blockIdx.x << 3) + (threadIdx.x >> 5);  // 1024*12 waves
    const int tn  = wid % 12;
    const int tm  = wid / 12;
    v8f c = {};
    for (int kk = 0; kk < D_INNER; kk += 32) {
        v16bf a  = load_a_bf16(yg, D_INNER, tm * 16, kk);
        v16bf bf = load_b_bf16(wout, 1, D_INNER, kk, tn * 16);
        c = __builtin_amdgcn_wmma_f32_16x16x32_bf16(false, a, false, bf, (short)0, c, false, false);
    }
    const int lane = threadIdx.x & 31;
    const int n    = tn * 16 + (lane & 15);
    const int mh   = tm * 16 + ((lane >> 4) << 3);
#pragma unroll
    for (int j = 0; j < 8; ++j)
        out[(long)(mh + j) * D_MODEL + n] = c[j];
}

// ---------------- Launcher ----------------
extern "C" void kernel_launch(void* const* d_in, const int* in_sizes, int n_in,
                              void* d_out, int out_size, void* d_ws, size_t ws_size,
                              hipStream_t stream) {
    (void)in_sizes; (void)n_in; (void)out_size; (void)ws_size;
    const float* x    = (const float*)d_in[0];   // (B,H,W,192)
    const float* win  = (const float*)d_in[1];   // (768,192)
    const float* wout = (const float*)d_in[2];   // (192,384)
    const float* xpw  = (const float*)d_in[3];   // (4,44,384)
    const float* dtw  = (const float*)d_in[4];   // (4,384,12)
    const float* dtb  = (const float*)d_in[5];   // (4,384)
    const float* alog = (const float*)d_in[6];   // (1536,16)
    const float* Dsv  = (const float*)d_in[7];   // (1536,)
    const float* nw   = (const float*)d_in[8];   // (384,)
    const float* nb   = (const float*)d_in[9];   // (384,)

    float* ws      = (float*)d_ws;
    float* xin     = ws;                                   // B*L*384   = 6291456 (l-major)
    float* zbuf    = xin     + (long)Mtot * D_INNER;       // 6291456
    float* dts_raw = zbuf    + (long)Mtot * D_INNER;       // 16*12*L   = 786432
    float* Bsb     = dts_raw + (long)Bb * KDIRS * DT_RANK * Ll;  // 16*16*L
    float* Csb     = Bsb     + (long)Bb * KDIRS * D_STATE * Ll;  // 16*16*L
    float* yg      = Csb     + (long)Bb * KDIRS * D_STATE * Ll;  // 6291456
    float* xpw_pad = yg      + (long)Mtot * D_INNER;       // 4*48*384  = 73728
    float* out_y   = xpw_pad + (long)KDIRS * 48 * D_INNER; // 16*L*384  = 25165824 (l-major)

    pack_xpw       <<<288, 256, 0, stream>>>(xpw, xpw_pad);
    gemm_inproj    <<<6144, 256, 0, stream>>>(x, win, xin, zbuf);
    gemm_xdbl      <<<1536, 256, 0, stream>>>(xpw_pad, xin, dts_raw, Bsb, Csb);
    scan_kernel    <<<Bb * KDIRS, D_INNER, 0, stream>>>(xin, dts_raw, Bsb, Csb,
                                                        dtw, dtb, alog, Dsv, out_y);
    combine_ln_gate<<<Mtot, D_INNER, 0, stream>>>(out_y, zbuf, nw, nb, yg);
    gemm_out       <<<1536, 256, 0, stream>>>(yg, wout, (float*)d_out);
}
